// IterBlock_63840393888311
// MI455X (gfx1250) — matrix operand order repaired and checked by reading, not compiled, not run on hardware
//
#include <hip/hip_runtime.h>
#include <hip/hip_bf16.h>

typedef __attribute__((ext_vector_type(16))) __bf16 v16bf;
typedef __attribute__((ext_vector_type(8)))  __bf16 v8bf;
typedef __attribute__((ext_vector_type(8)))  float  v8f;
typedef __attribute__((ext_vector_type(4)))  float  f32x4;

#define IMG    256
#define NPIX   (IMG * IMG)
#define GRIDP  126
#define NPATCH (GRIDP * GRIDP)   // 15876
#define PP     36
#define HID    64
#define TILE_M 128
#define TILE_K 32
#define NTILES 4                 // 4 * 16 = 64 output columns per wave row-tile

// ---------------- conv1: 1 -> 64, 3x3 SAME, relu ----------------
__global__ void conv1_kernel(const float* __restrict__ in,
                             const float* __restrict__ w1,
                             const float* __restrict__ b1,
                             float* __restrict__ h1) {
    int p = blockIdx.x * blockDim.x + threadIdx.x;
    if (p >= NPIX) return;
    int y = p >> 8, x = p & 255;
    float pix[9];
#pragma unroll
    for (int dy = 0; dy < 3; ++dy)
#pragma unroll
        for (int dx = 0; dx < 3; ++dx) {
            int yy = y + dy - 1, xx = x + dx - 1;
            pix[dy * 3 + dx] = (yy >= 0 && yy < IMG && xx >= 0 && xx < IMG)
                                   ? in[yy * IMG + xx] : 0.f;
        }
    for (int co = 0; co < 64; ++co) {
        float acc = b1[co];
#pragma unroll
        for (int k = 0; k < 9; ++k) acc += w1[co * 9 + k] * pix[k];
        h1[co * NPIX + p] = fmaxf(acc, 0.f);
    }
}

// ---------------- conv2: 64 -> 64, 3x3 SAME, relu (LDS tiled) ----------------
__global__ __launch_bounds__(256) void conv2_kernel(const float* __restrict__ h1,
                                                    const float* __restrict__ w2,
                                                    const float* __restrict__ b2,
                                                    float* __restrict__ h2) {
    __shared__ float tile[64 * 10 * 10];          // 25.6 KB: 8x8 pixel tile + halo, all 64 ci
    int tid = threadIdx.x;
    int bx = blockIdx.x, by = blockIdx.y;
    for (int idx = tid; idx < 6400; idx += 256) {
        int ch = idx / 100;
        int rr = (idx % 100) / 10;
        int cc = idx % 10;
        int gy = by * 8 + rr - 1, gx = bx * 8 + cc - 1;
        tile[idx] = (gy >= 0 && gy < IMG && gx >= 0 && gx < IMG)
                        ? h1[ch * NPIX + gy * IMG + gx] : 0.f;
    }
    __syncthreads();
    int pix = tid & 63;        // 8x8 pixels
    int cog = tid >> 6;        // 4 groups of 16 output channels
    int ly = pix >> 3, lx = pix & 7;
    float acc[16];
#pragma unroll
    for (int c = 0; c < 16; ++c) acc[c] = b2[cog * 16 + c];
    for (int ci = 0; ci < 64; ++ci) {
        float p[9];
        int base = ci * 100 + ly * 10 + lx;
#pragma unroll
        for (int dy = 0; dy < 3; ++dy)
#pragma unroll
            for (int dx = 0; dx < 3; ++dx)
                p[dy * 3 + dx] = tile[base + dy * 10 + dx];
#pragma unroll
        for (int c = 0; c < 16; ++c) {
            int wb = ((cog * 16 + c) * 64 + ci) * 9;
            float a = acc[c];
#pragma unroll
            for (int k = 0; k < 9; ++k) a += w2[wb + k] * p[k];
            acc[c] = a;
        }
    }
    int y = by * 8 + ly, x = bx * 8 + lx;
#pragma unroll
    for (int c = 0; c < 16; ++c)
        h2[(cog * 16 + c) * NPIX + y * IMG + x] = fmaxf(acc[c], 0.f);
}

// ---------------- conv3: 64 -> 1, 3x3 SAME (no relu) ----------------
__global__ void conv3_kernel(const float* __restrict__ h2,
                             const float* __restrict__ w3,
                             const float* __restrict__ b3,
                             float* __restrict__ tmp2) {
    int p = blockIdx.x * blockDim.x + threadIdx.x;
    if (p >= NPIX) return;
    int y = p >> 8, x = p & 255;
    float acc = b3[0];
    for (int ci = 0; ci < 64; ++ci) {
#pragma unroll
        for (int dy = 0; dy < 3; ++dy)
#pragma unroll
            for (int dx = 0; dx < 3; ++dx) {
                int yy = y + dy - 1, xx = x + dx - 1;
                if (yy >= 0 && yy < IMG && xx >= 0 && xx < IMG)
                    acc += w3[ci * 9 + dy * 3 + dx] * h2[ci * NPIX + yy * IMG + xx];
            }
    }
    tmp2[p] = acc;
}

// ---------------- patch extraction: [NPATCH, 36] ----------------
__global__ void patch_kernel(const float* __restrict__ in, float* __restrict__ Xp) {
    int t = blockIdx.x * blockDim.x + threadIdx.x;
    if (t >= NPATCH * PP) return;
    int p = t / PP, e = t % PP;
    int pi = p / GRIDP, pj = p % GRIDP;
    int i = e / 6, j = e % 6;
    Xp[t] = in[(2 * pi + i) * IMG + (2 * pj + j)];
}

// ---------------- skinny projection: S = X @ W  ([N,fin] @ [fin,fout]) ----------------
__global__ void gcn_in_kernel(const float* __restrict__ X, const float* __restrict__ w,
                              float* __restrict__ S, int fin, int fout) {
    int t = blockIdx.x * blockDim.x + threadIdx.x;
    if (t >= NPATCH * fout) return;
    int p = t / fout, o = t % fout;
    float acc = 0.f;
    for (int f = 0; f < fin; ++f) acc += X[p * fin + f] * w[f * fout + o];
    S[t] = acc;
}

// ---- staging helpers for the big GEMM (branch-free fast path; uniform tail path) ----
// B indices fit in 32 bits (K*ncols < 2^21), so use int math: no v_mul_u64 in the loop.
__device__ __forceinline__ void gemm_stage_load(const float* __restrict__ A,
                                                const float* __restrict__ B,
                                                size_t aoff, int ahalf, int bq, int bnc,
                                                int K, int ncols, int k0,
                                                float* ar, float* br) {
    if (k0 + TILE_K <= K) {
        // fast path: 4x global_load_b128 for A chunk, 8x b32 (L2-resident) for B
        const f32x4* ap = (const f32x4*)(A + aoff + k0 + ahalf);
#pragma unroll
        for (int q = 0; q < 4; ++q) {
            f32x4 v = ap[q];
#pragma unroll
            for (int e = 0; e < 4; ++e) ar[q * 4 + e] = v[e];
        }
        int boff = (k0 + bq) * ncols + bnc;     // 32-bit index math
#pragma unroll
        for (int e = 0; e < 8; ++e)
            br[e] = B[boff + e * ncols];
    } else {
        // K-tail (uniform across the grid: last kstep only)
#pragma unroll
        for (int e = 0; e < 16; ++e) {
            int gk = k0 + ahalf + e;
            ar[e] = (gk < K) ? A[aoff + gk] : 0.f;
        }
#pragma unroll
        for (int e = 0; e < 8; ++e) {
            int gk = k0 + bq + e;
            br[e] = (gk < K) ? B[gk * ncols + bnc] : 0.f;
        }
    }
}

__device__ __forceinline__ void gemm_stage_store(__bf16* lsA, __bf16* lsB,
                                                 int arow, int ahalf, int bn, int bq,
                                                 const float* ar, const float* br) {
    v16bf av;
#pragma unroll
    for (int e = 0; e < 16; ++e) av[e] = (__bf16)ar[e];
    *(v16bf*)(lsA + arow * TILE_K + ahalf) = av;     // 32B-aligned -> 2x ds_store_b128
    v8bf bv;
#pragma unroll
    for (int e = 0; e < 8; ++e) bv[e] = (__bf16)br[e];
    *(v8bf*)(lsB + bn * TILE_K + bq) = bv;           // 16B-aligned -> ds_store_b128
}

// ---------------- big GEMM: C = act(adj @ B + bias) via v_wmma bf16 ----------------
// A: [M,K] f32 row-major (adj), B: [K,ncols] f32, C: [M,ncols] f32
// 256 threads = 8 waves; block tile 128x64; double-buffered LDS, 1 barrier per K-step.
__global__ __launch_bounds__(256) void gcn_big_gemm(const float* __restrict__ A,
                                                    const float* __restrict__ B,
                                                    const float* __restrict__ bias,
                                                    float* __restrict__ C,
                                                    int M, int K, int ncols, int relu) {
    __shared__ __bf16 lsA[2][TILE_M * TILE_K];   // 2 x 8 KB
    __shared__ __bf16 lsB[2][64 * TILE_K];       // 2 x 4 KB  (transposed [n][k])
    const int tid  = threadIdx.x;
    const int wave = tid >> 5;
    const int lane = tid & 31;
    const int m0   = blockIdx.x * TILE_M;

    // staging coordinates
    const int arow  = tid >> 1;            // 0..127
    const int ahalf = (tid & 1) * 16;      // 0 or 16
    int gmrow = m0 + arow;  if (gmrow >= M) gmrow = M - 1;   // clamp: feeds masked C rows only
    const size_t aoff = (size_t)gmrow * (size_t)K;
    const int bn  = tid & 63;              // lsB row (output column)
    const int bq  = (tid >> 6) * 8;        // k sub-chunk
    const int bnc = (bn < ncols) ? bn : (ncols - 1);         // clamp: feeds masked C cols only

    const int ksteps = (K + TILE_K - 1) / TILE_K;

    float ar[16], br[8];

    // prologue: stage tile 0 into buffer 0
    gemm_stage_load(A, B, aoff, ahalf, bq, bnc, K, ncols, 0, ar, br);
    gemm_stage_store(lsA[0], lsB[0], arow, ahalf, bn, bq, ar, br);
    __syncthreads();

    v8f acc[NTILES];
#pragma unroll
    for (int t = 0; t < NTILES; ++t) acc[t] = (v8f){0.f,0.f,0.f,0.f,0.f,0.f,0.f,0.f};

    // WMMA fragment coordinates (ISA 7.12.2 layouts)
    const int mrow  = wave * 16 + (lane & 15);
    const int kbase = (lane < 16) ? 0 : 8;
    const int nl    = lane & 15;
    const int kb2   = (lane < 16) ? 0 : 16;

    for (int ks = 0; ks < ksteps; ++ks) {
        const int cur = ks & 1;
        const bool have_next = (ks + 1 < ksteps);
        // issue next tile's global loads first: latency overlaps the WMMA work below
        if (have_next)
            gemm_stage_load(A, B, aoff, ahalf, bq, bnc, K, ncols, (ks + 1) * TILE_K, ar, br);

        const __bf16* la = lsA[cur];
        const __bf16* lb = lsB[cur];
        v8bf alo = *(const v8bf*)(la + mrow * TILE_K + kbase);
        v8bf ahi = *(const v8bf*)(la + mrow * TILE_K + 16 + kbase);
        v16bf afrag = __builtin_shufflevector(alo, ahi,
                                              0,1,2,3,4,5,6,7,8,9,10,11,12,13,14,15);
        // load ALL B fragments first (distinct regs) so the ds_loads batch up and the
        // four WMMAs can issue back-to-back with partial dscnt waits
        v16bf bfrag[NTILES];
#pragma unroll
        for (int t = 0; t < NTILES; ++t)
            bfrag[t] = __builtin_shufflevector(*(const v8bf*)(lb + (t * 16 + nl) * TILE_K + kbase), *(const v8bf*)(lb + (t * 16 + nl) * TILE_K + 16 + kbase), 0,1,2,3,4,5,6,7,8,9,10,11,12,13,14,15);
#pragma unroll
        for (int t = 0; t < NTILES; ++t)
            acc[t] = __builtin_amdgcn_wmma_f32_16x16x32_bf16(
                false, afrag, false, bfrag[t], (short)0, acc[t], false, false);

        if (have_next) {
            // safe: last reads of the other buffer happened before the previous barrier
            gemm_stage_store(lsA[cur ^ 1], lsB[cur ^ 1], arow, ahalf, bn, bq, ar, br);
            __syncthreads();
        }
    }

    // C/D layout: lane L, vgpr r -> m = r + 8*(L>=16), n = L%16
    const int mloc = wave * 16 + ((lane >= 16) ? 8 : 0);
    const int nloc = lane & 15;
    for (int t = 0; t < NTILES; ++t) {
        int n = t * 16 + nloc;
        if (n >= ncols) continue;
        float bv = bias[n];
#pragma unroll
        for (int r = 0; r < 8; ++r) {
            int gm = m0 + mloc + r;
            if (gm < M) {
                float v = acc[t][r] + bv;
                if (relu) v = fmaxf(v, 0.f);
                C[(size_t)gm * ncols + n] = v;
            }
        }
    }
}

// ---------------- final: relu(tmp1 + tmp2 + fold(T2)/count) ----------------
__global__ void final_kernel(const float* __restrict__ in, const float* __restrict__ proj,
                             const float* __restrict__ pw, const float* __restrict__ tmp2,
                             const float* __restrict__ T2, float* __restrict__ out) {
    int p = blockIdx.x * blockDim.x + threadIdx.x;
    if (p >= NPIX) return;
    int y = p >> 8, x = p & 255;
    float sum = 0.f;
    int cnt = 0;
#pragma unroll
    for (int i = 0; i < 6; ++i) {
        int r = y - i;
        if (r < 0 || r > 250 || (r & 1)) continue;
#pragma unroll
        for (int j = 0; j < 6; ++j) {
            int c = x - j;
            if (c < 0 || c > 250 || (c & 1)) continue;
            int pp = (r >> 1) * GRIDP + (c >> 1);
            sum += T2[pp * PP + i * 6 + j];
            cnt++;
        }
    }
    float t3 = sum / (float)cnt;
    float a  = in[p];
    float t1 = a - pw[0] * (a - proj[p]);
    out[p] = fmaxf(t1 + tmp2[p] + t3, 0.f);
}

extern "C" void kernel_launch(void* const* d_in, const int* in_sizes, int n_in,
                              void* d_out, int out_size, void* d_ws, size_t ws_size,
                              hipStream_t stream) {
    (void)in_sizes; (void)n_in; (void)out_size; (void)ws_size;
    const float* input   = (const float*)d_in[0];
    const float* proj    = (const float*)d_in[1];
    const float* adj     = (const float*)d_in[2];
    const float* prj_w   = (const float*)d_in[3];
    const float* conv_w1 = (const float*)d_in[4];
    const float* conv_b1 = (const float*)d_in[5];
    const float* conv_w2 = (const float*)d_in[6];
    const float* conv_b2 = (const float*)d_in[7];
    const float* conv_w3 = (const float*)d_in[8];
    const float* conv_b3 = (const float*)d_in[9];
    const float* gcn3_w  = (const float*)d_in[10];
    const float* gcn3_b  = (const float*)d_in[11];
    const float* gcn4_w  = (const float*)d_in[12];
    const float* gcn4_b  = (const float*)d_in[13];
    float* out = (float*)d_out;

    char* ws = (char*)d_ws;
    size_t off = 0;
    auto alloc = [&](size_t bytes) {
        float* r = (float*)(ws + off);
        off += (bytes + 255) & ~(size_t)255;
        return r;
    };
    float* h1   = alloc((size_t)64 * NPIX * 4);     // 16.78 MB
    float* h2   = alloc((size_t)64 * NPIX * 4);     // 16.78 MB
    float* tmp2 = alloc((size_t)NPIX * 4);
    float* Xp   = alloc((size_t)NPATCH * PP * 4);
    float* S1   = alloc((size_t)NPATCH * HID * 4);
    float* T1   = alloc((size_t)NPATCH * HID * 4);
    float* S2   = alloc((size_t)NPATCH * PP * 4);
    float* T2   = alloc((size_t)NPATCH * PP * 4);

    conv1_kernel<<<NPIX / 256, 256, 0, stream>>>(input, conv_w1, conv_b1, h1);
    conv2_kernel<<<dim3(32, 32), 256, 0, stream>>>(h1, conv_w2, conv_b2, h2);
    conv3_kernel<<<NPIX / 256, 256, 0, stream>>>(h2, conv_w3, conv_b3, tmp2);

    patch_kernel<<<(NPATCH * PP + 255) / 256, 256, 0, stream>>>(input, Xp);
    gcn_in_kernel<<<(NPATCH * HID + 255) / 256, 256, 0, stream>>>(Xp, gcn3_w, S1, PP, HID);
    gcn_big_gemm<<<(NPATCH + TILE_M - 1) / TILE_M, 256, 0, stream>>>(
        adj, S1, gcn3_b, T1, NPATCH, NPATCH, HID, 1);
    gcn_in_kernel<<<(NPATCH * PP + 255) / 256, 256, 0, stream>>>(T1, gcn4_w, S2, HID, PP);
    gcn_big_gemm<<<(NPATCH + TILE_M - 1) / TILE_M, 256, 0, stream>>>(
        adj, S2, gcn4_b, T2, NPATCH, NPATCH, PP, 0);

    final_kernel<<<NPIX / 256, 256, 0, stream>>>(input, proj, prj_w, tmp2, T2, out);
}
